// Codebook_57861799412438
// MI455X (gfx1250) — matrix operand order repaired and checked by reading, not compiled, not run on hardware
//
#include <hip/hip_runtime.h>

typedef __bf16 bf16;
typedef __attribute__((ext_vector_type(4)))  float        f32x4;
typedef __attribute__((ext_vector_type(8)))  float        v8f;
typedef __attribute__((ext_vector_type(8)))  bf16         bf16x8;
typedef __attribute__((ext_vector_type(16))) bf16         v16bf;
typedef __attribute__((ext_vector_type(4)))  unsigned int u32x4;
typedef __attribute__((ext_vector_type(4)))  int          i32x4;
typedef __attribute__((ext_vector_type(8)))  int          i32x8;

#define D_DIM    256
#define K_CODES  1024
#define TILE_M   32
#define CHUNK    64
#define NCHUNKS  (K_CODES / CHUNK)
#define LSTRIDE  264                 // 256 + 8 bf16 pad (TDM pads 4 DWORDs / 128 DWORDs)
#define EB_ELEMS (CHUNK * LSTRIDE)   // 16896 bf16 per chunk buffer

// ---- Tensor Data Mover: toolchain-portable call (5-arg ROCm7.2 / 6-arg therock) ----
#if __has_include(<hip/amd_detail/amd_gfx1250_TDM.h>)
#define TDM_TENSOR_LOAD(g0, g1, z4, z8) \
    __builtin_amdgcn_tensor_load_to_lds((g0), (g1), (z4), (z4), (z8), 0)
#else
#define TDM_TENSOR_LOAD(g0, g1, z4, z8) \
    __builtin_amdgcn_tensor_load_to_lds((g0), (g1), (z4), (z4), 0)
#endif

// Load one 64x256 bf16 tile (row-major, 512B rows) into LDS, padding each row
// by 4 DWORDs so the LDS row stride is LSTRIDE bf16 elements.
__device__ __forceinline__ void tdm_load_chunk(unsigned lds_byte_addr, const bf16* gsrc) {
    unsigned long long ga = (unsigned long long)(uintptr_t)gsrc;
    u32x4 g0;
    g0[0] = 1u;                                            // count = 1 valid descriptor
    g0[1] = lds_byte_addr;                                 // lds_addr
    g0[2] = (unsigned)ga;                                  // global_addr[31:0]
    g0[3] = ((unsigned)(ga >> 32) & 0x01FFFFFFu) | (2u << 30); // addr[56:32] | type=2
    i32x8 g1;
    g1[0] = (int)((1u << 16)      // data_size = 2 bytes
                | (1u << 20)      // pad_enable
                | (6u << 22)      // pad_interval: 128 DWORDs (one 256-elem bf16 row)
                | (3u << 25));    // pad_amount: 4 DWORDs (8 bf16)
    g1[1] = (int)(256u << 16);    // tensor_dim0 = 256 elements (bits 79:48 low half)
    g1[2] = (int)(1024u << 16);   // tensor_dim1 = 1024 rows
    g1[3] = (int)(256u << 16);    // tile_dim0 = 256
    g1[4] = (int)CHUNK;           // tile_dim1 = 64, tile_dim2 = 0
    g1[5] = 256;                  // tensor_dim0_stride = 256 elements
    g1[6] = 0;
    g1[7] = 0;
    i32x4 z4 = (i32x4){0, 0, 0, 0};
    i32x8 z8 = (i32x8){0, 0, 0, 0, 0, 0, 0, 0};
    (void)z8;
    TDM_TENSOR_LOAD(g0, g1, z4, z8);
}

// -------- kernel 1: split codebook to bf16 hi/lo + ||e_k||^2 --------
__global__ __launch_bounds__(256)
void vq_prep_kernel(const float* __restrict__ cb, bf16* __restrict__ cbHi,
                    bf16* __restrict__ cbLo, float* __restrict__ enorm) {
    __shared__ float red[256];
    const int k = blockIdx.x;
    const int d = threadIdx.x;
    float f = cb[(size_t)k * D_DIM + d];
    bf16 h = (bf16)f;
    bf16 l = (bf16)(f - (float)h);
    cbHi[(size_t)k * D_DIM + d] = h;
    cbLo[(size_t)k * D_DIM + d] = l;
    red[d] = f * f;
    __syncthreads();
    #pragma unroll
    for (int s = 128; s > 0; s >>= 1) {
        if (d < s) red[d] += red[d + s];
        __syncthreads();
    }
    if (d == 0) enorm[k] = red[0];
}

// -------- kernel 2: WMMA distances + argmin (TDM double-buffered codebook) --------
__global__ __launch_bounds__(256)
void vq_argmin_kernel(const float* __restrict__ z,
                      const bf16* __restrict__ cbHi, const bf16* __restrict__ cbLo,
                      const float* __restrict__ enorm,
                      float* __restrict__ idx_out, int* __restrict__ code_out) {
    // Layout: buf0 hi | buf0 lo | buf1 hi | buf1 lo. The z staging area aliases
    // buf1-hi: it is consumed into registers before any TDM targets buf1.
    __shared__ __align__(16) bf16 smem[4 * EB_ELEMS];
    __shared__ unsigned long long bestK[TILE_M];

    const int t    = threadIdx.x;
    const int lane = t & 31;
    const int wave = t >> 5;
    const int n0   = blockIdx.x * TILE_M;
    const int b    = n0 >> 12;
    const int hw0  = n0 & 4095;

    const unsigned smemBase = (unsigned)(uintptr_t)(&smem[0]);

    if (t < TILE_M) bestK[t] = ~0ull;

    // Prefetch chunk 0 via TDM while we stage z (buf0 is not used for staging).
    if (wave == 0) {
        tdm_load_chunk(smemBase + 0u * EB_ELEMS * 2u, cbHi);
        tdm_load_chunk(smemBase + 1u * EB_ELEMS * 2u, cbLo);
    }

    bf16* zAhi = smem + 2 * EB_ELEMS;
    bf16* zAlo = smem + 2 * EB_ELEMS + TILE_M * LSTRIDE;

    // ---- stage z tile, split into bf16 hi/lo (rows = n, cols = channel) ----
    {
        const int c = t;   // one channel per thread; 32 contiguous spatial floats
        const float* src = z + (((size_t)(b * 256 + c)) << 12) + hw0;
        #pragma unroll
        for (int i4 = 0; i4 < TILE_M / 4; ++i4) {
            f32x4 v = ((const f32x4*)src)[i4];
            #pragma unroll
            for (int j = 0; j < 4; ++j) {
                int n = i4 * 4 + j;
                float f = v[j];
                bf16 h = (bf16)f;
                bf16 l = (bf16)(f - (float)h);
                zAhi[n * LSTRIDE + c] = h;
                zAlo[n * LSTRIDE + c] = l;
            }
        }
    }
    __syncthreads();

    // wave -> output tile mapping: 2 row tiles x 4 col tiles
    const int rowA    = (wave & 1) * 16 + (lane & 15);
    const int colTile = wave >> 1;
    const int rowB    = colTile * 16 + (lane & 15);
    const int aoff    = (lane & 16) ? 8 : 0;    // ISA 16-bit A layout
    const int boff    = (lane & 16) ? 8 : 0;    // ISA 16-bit B layout

    // ---- load A fragments (hi+lo) into registers once ----
    v16bf aH[8], aL[8];
    #pragma unroll
    for (int ks = 0; ks < 8; ++ks) {
        const bf16* ph = zAhi + rowA * LSTRIDE + ks * 32 + aoff;
        const bf16* pl = zAlo + rowA * LSTRIDE + ks * 32 + aoff;
        bf16x8 h0 = *(const bf16x8*)ph;
        bf16x8 h1 = *(const bf16x8*)(ph + 16);
        bf16x8 l0 = *(const bf16x8*)pl;
        bf16x8 l1 = *(const bf16x8*)(pl + 16);
        aH[ks] = __builtin_shufflevector(h0, h1, 0,1,2,3,4,5,6,7,8,9,10,11,12,13,14,15);
        aL[ks] = __builtin_shufflevector(l0, l1, 0,1,2,3,4,5,6,7,8,9,10,11,12,13,14,15);
    }
    __syncthreads();                                  // staging consumed; buf1 free
    if (wave == 0) __builtin_amdgcn_s_wait_tensorcnt(0);
    __syncthreads();                                  // chunk 0 visible to all

    float bestD[8];
    int   bestC[8];
    #pragma unroll
    for (int v = 0; v < 8; ++v) { bestD[v] = 3.0e38f; bestC[v] = 0; }

    for (int ch = 0; ch < NCHUNKS; ++ch) {
        const int cur = ch & 1;

        // Issue next chunk's TDM loads into the other buffer (overlaps compute).
        if (wave == 0 && ch + 1 < NCHUNKS) {
            const size_t goff = (size_t)(ch + 1) * CHUNK * D_DIM;
            const unsigned dstH = smemBase + (unsigned)(((1 - cur) * 2 + 0) * EB_ELEMS * 2);
            const unsigned dstL = smemBase + (unsigned)(((1 - cur) * 2 + 1) * EB_ELEMS * 2);
            tdm_load_chunk(dstH, cbHi + goff);
            tdm_load_chunk(dstL, cbLo + goff);
        }

        const bf16* ebh = smem + (cur * 2 + 0) * EB_ELEMS;
        const bf16* ebl = smem + (cur * 2 + 1) * EB_ELEMS;

        v8f acc = {0.f, 0.f, 0.f, 0.f, 0.f, 0.f, 0.f, 0.f};
        #pragma unroll
        for (int ks = 0; ks < 8; ++ks) {
            const int k0 = ks * 32;
            const bf16* pbh = ebh + rowB * LSTRIDE + k0 + boff;
            const bf16* pbl = ebl + rowB * LSTRIDE + k0 + boff;
            bf16x8 b0h = *(const bf16x8*)pbh;
            bf16x8 b1h = *(const bf16x8*)(pbh + 16);
            bf16x8 b0l = *(const bf16x8*)pbl;
            bf16x8 b1l = *(const bf16x8*)(pbl + 16);
            v16bf bh = __builtin_shufflevector(b0h, b1h, 0,1,2,3,4,5,6,7,8,9,10,11,12,13,14,15);
            v16bf bl = __builtin_shufflevector(b0l, b1l, 0,1,2,3,4,5,6,7,8,9,10,11,12,13,14,15);
            // 3-term bf16 split ~= fp32 dot product, f32 accumulate
            acc = __builtin_amdgcn_wmma_f32_16x16x32_bf16(false, aH[ks], false, bh, (short)0, acc, false, false);
            acc = __builtin_amdgcn_wmma_f32_16x16x32_bf16(false, aL[ks], false, bh, (short)0, acc, false, false);
            acc = __builtin_amdgcn_wmma_f32_16x16x32_bf16(false, aH[ks], false, bl, (short)0, acc, false, false);
        }

        const int code = ch * CHUNK + colTile * 16 + (lane & 15);
        const float en = enorm[code];
        #pragma unroll
        for (int v = 0; v < 8; ++v) {
            float s = en - 2.0f * acc[v];   // ||z||^2 dropped: constant per row
            if (s < bestD[v]) { bestD[v] = s; bestC[v] = code; }
        }

        if (wave == 0) __builtin_amdgcn_s_wait_tensorcnt(0);
        __syncthreads();
    }

    // ---- cross-lane/wave reduction via LDS u64 atomic min ----
    const int rowBase = (wave & 1) * 16 + ((lane & 16) ? 8 : 0);
    #pragma unroll
    for (int v = 0; v < 8; ++v) {
        int row = rowBase + v;
        int ib = __float_as_int(bestD[v]);
        unsigned u = (ib >= 0) ? ((unsigned)ib | 0x80000000u) : ~(unsigned)ib;
        unsigned long long key = ((unsigned long long)u << 32) | (unsigned)bestC[v];
        atomicMin(&bestK[row], key);   // ties -> lowest code, matches argmin
    }
    __syncthreads();
    if (t < TILE_M) {
        unsigned code = (unsigned)(bestK[t] & 0xFFFFFFFFull);
        idx_out[n0 + t]  = (float)code;
        code_out[n0 + t] = (int)code;
    }
}

// -------- kernel 3: z_q gather (NCHW) + loss accumulation --------
__global__ __launch_bounds__(256)
void vq_zq_loss_kernel(const float* __restrict__ z, const float* __restrict__ cb,
                       const int* __restrict__ codes, float* __restrict__ zq,
                       float* __restrict__ lossAcc) {
    __shared__ int   cds[64];
    __shared__ float red[256];
    const int t   = threadIdx.x;
    const int n0  = blockIdx.x * 64;
    const int b   = n0 >> 12;
    const int hw0 = n0 & 4095;
    if (t < 64) cds[t] = codes[n0 + t];
    __syncthreads();
    const int nl     = t & 63;
    const int csub   = t >> 6;
    const int myCode = cds[nl];
    float acc = 0.f;
    #pragma unroll 4
    for (int co = 0; co < 64; ++co) {
        int c = co * 4 + csub;
        size_t off = (((size_t)(b * 256 + c)) << 12) + hw0 + nl;
        float zv = z[off];
        float q  = cb[(size_t)myCode * D_DIM + c];
        zq[off] = q;                     // straight-through forward value == z_q
        float d = q - zv;
        acc += d * d;
    }
    red[t] = acc;
    __syncthreads();
    #pragma unroll
    for (int s = 128; s > 0; s >>= 1) {
        if (t < s) red[t] += red[t + s];
        __syncthreads();
    }
    if (t == 0) atomicAdd(lossAcc, red[0]);
}

// -------- kernel 4: scale loss --------
__global__ void vq_loss_scale_kernel(float* loss) {
    if (threadIdx.x == 0 && blockIdx.x == 0)
        *loss *= 1.25f / 16777216.0f;   // (1 + beta) * mean
}

extern "C" void kernel_launch(void* const* d_in, const int* in_sizes, int n_in,
                              void* d_out, int out_size, void* d_ws, size_t ws_size,
                              hipStream_t stream) {
    const float* z  = (const float*)d_in[0];   // [16,256,64,64]
    const float* cb = (const float*)d_in[1];   // [1024,256]
    float* out   = (float*)d_out;
    float* zq    = out;                                   // 16777216 floats
    float* idx_f = out + (size_t)16 * 256 * 64 * 64;      // 65536 floats
    float* loss  = idx_f + 65536;                         // 1 float

    float* enorm = (float*)d_ws;                          // 1024 floats @ 0
    int*   codes = (int*)((char*)d_ws + 4096);            // 65536 ints  @ 4 KB
    bf16*  cbHi  = (bf16*)((char*)d_ws + 266240);         // 512 KB
    bf16*  cbLo  = (bf16*)((char*)d_ws + 790528);         // 512 KB

    hipMemsetAsync(loss, 0, sizeof(float), stream);
    vq_prep_kernel<<<K_CODES, 256, 0, stream>>>(cb, cbHi, cbLo, enorm);
    vq_argmin_kernel<<<65536 / TILE_M, 256, 0, stream>>>(z, cbHi, cbLo, enorm, idx_f, codes);
    vq_zq_loss_kernel<<<65536 / 64, 256, 0, stream>>>(z, cb, codes, zq, loss);
    vq_loss_scale_kernel<<<1, 1, 0, stream>>>(loss);
}
